// KANLinear_60610578481557
// MI455X (gfx1250) — matrix operand order repaired and checked by reading, not compiled, not run on hardware
//
#include <hip/hip_runtime.h>
#include <hip/hip_bf16.h>

// KAN linear:  out[b,o] = sum_{i,d} basis(tanh(x[b,i]))[d] * coef[i,o,d]
// == GEMM  A[16384,256](f16) x B[256,4096](f16) -> C[16384,4096](f32)
// Store-bound (256MB out @ 23.3TB/s ~= 11us); f16 WMMA keeps compute under that.

typedef __attribute__((ext_vector_type(16))) _Float16 v16h;
typedef __attribute__((ext_vector_type(8)))  float    v8f;

#define BATCH   16384
#define INF     16
#define OUTF    4096
#define KDIM    256      // INF * 16 basis values
#define NKNOTS  32
#define DEG     15

// ---------------------------------------------------------------------------
// Kernel 1: tanh + uniform-knot Cox-de-Boor basis, packed into WMMA A-fragment
// order.  A-fragment (16-bit A 16x32, ISA 7.12.2): lane = g*16 + (m%16),
//   g=0: K in {0..15}(e=K)   (contiguous k order, same as B; not the ISA table)
//   g=1: K in {16..31}(e=K-16)
// packedA index (in halves) = ((mTile*8 + kStep)*32 + lane)*16 + e
// ---------------------------------------------------------------------------
__global__ void kan_basis_pack(const float* __restrict__ x,
                               _Float16* __restrict__ pA) {
    const int tid = blockIdx.x * blockDim.x + threadIdx.x;   // b*16 + i
    const int b = tid >> 4;
    const int i = tid & 15;

    const float xv   = tanhf(x[tid]);
    const float step = 2.0f / 31.0f;

    float B[31];
#pragma unroll
    for (int j = 0; j < 31; ++j) {
        const float k0 = -1.0f + (float)j * step;
        const float k1 = -1.0f + (float)(j + 1) * step;
        B[j] = (k0 <= xv && xv < k1) ? 1.0f : 0.0f;
    }
    // uniform knots: every denominator at level k is k*step
#pragma unroll
    for (int k = 1; k <= DEG; ++k) {
        const float inv = 31.0f / (2.0f * (float)k);
#pragma unroll
        for (int j = 0; j < 31 - k; ++j) {
            const float tj   = -1.0f + (float)j * step;
            const float tk1j = -1.0f + (float)(k + 1 + j) * step;
            const float w1 = (xv - tj) * inv;
            const float w2 = (tk1j - xv) * inv;
            B[j] = w1 * B[j] + w2 * B[j + 1];
        }
    }

    const long base = ((long)(b >> 4) * 8 + (i >> 1)) * 32;  // (mTile*8 + kStep)*32
    const int  r    = b & 15;
#pragma unroll
    for (int d = 0; d < 16; ++d) {
        const int c = ((i & 1) << 4) + d;    // K within 32-wide k-step
        int g, e;
        g = c >> 4; e = c & 15;
        pA[(base + (long)(g * 16 + r)) * 16 + e] = (_Float16)B[d];
    }
}

// ---------------------------------------------------------------------------
// Kernel 2: coef[i,o,d] (f32) -> packed f16 B-fragments.
// B-fragment (16-bit B 32x16, contiguous k order here): lane = g*16 + (n%16); VGPR v halves (h) hold
// rows K = 2v + h + 16g  ->  element index e = K & 15 = d,  g = i&1, kStep=i>>1.
// packedB index (halves) = ((nTile*8 + kStep)*32 + g*16 + (o&15))*16 + d
// One thread per (i,o): 64B coalesced read, one 32B v16h write.
// ---------------------------------------------------------------------------
__global__ void kan_coef_pack(const float* __restrict__ coef,
                              _Float16* __restrict__ pB) {
    const int tid = blockIdx.x * blockDim.x + threadIdx.x;   // i*4096 + o
    const int i = tid >> 12;
    const int o = tid & 4095;

    const float4* src = reinterpret_cast<const float4*>(coef + (long)tid * 16);
    float v[16];
#pragma unroll
    for (int q = 0; q < 4; ++q) {
        const float4 f = src[q];
        v[q * 4 + 0] = f.x; v[q * 4 + 1] = f.y;
        v[q * 4 + 2] = f.z; v[q * 4 + 3] = f.w;
    }

    const long base = (((long)(o >> 4) * 8 + (i >> 1)) * 32
                       + ((i & 1) << 4) + (o & 15)) * 16;
    v16h hv;
#pragma unroll
    for (int d = 0; d < 16; ++d) hv[d] = (_Float16)v[d];
    *reinterpret_cast<v16h*>(pB + base) = hv;
}

// ---------------------------------------------------------------------------
// Kernel 3: GEMM.  One wave -> 16x64 output strip.  K=256 in 8 steps of 32.
// A fragment reused across 4 WMMAs per step -> 32 v_wmma_f32_16x16x32_f16.
// C layout (f32 16x16): vgpr p, lane L -> row p + 8*(L/16), col L%16.
// NT stores: 256MB output stream must not evict L2-resident A/B.
// ---------------------------------------------------------------------------
__global__ void kan_gemm(const _Float16* __restrict__ pA,
                         const _Float16* __restrict__ pB,
                         float* __restrict__ out) {
    const int wave = (blockIdx.x * blockDim.x + threadIdx.x) >> 5;
    const int lane = threadIdx.x & 31;
    const int nGroup = wave & 63;     // 64 column-groups of 64
    const int mTile  = wave >> 6;     // 1024 row tiles of 16

    v8f acc0 = {}, acc1 = {}, acc2 = {}, acc3 = {};

    const v16h* aBase = reinterpret_cast<const v16h*>(pA)
                        + (long)mTile * 8 * 32 + lane;      // + kStep*32
    const v16h* bBase = reinterpret_cast<const v16h*>(pB);
    const long  nT0   = (long)nGroup * 4;

#pragma unroll
    for (int ks = 0; ks < 8; ++ks) {
        const v16h a  = aBase[(long)ks * 32];
        const v16h b0 = bBase[((nT0 + 0) * 8 + ks) * 32 + lane];
        const v16h b1 = bBase[((nT0 + 1) * 8 + ks) * 32 + lane];
        const v16h b2 = bBase[((nT0 + 2) * 8 + ks) * 32 + lane];
        const v16h b3 = bBase[((nT0 + 3) * 8 + ks) * 32 + lane];
        acc0 = __builtin_amdgcn_wmma_f32_16x16x32_f16(false, a, false, b0,
                                                      (short)0, acc0, false, false);
        acc1 = __builtin_amdgcn_wmma_f32_16x16x32_f16(false, a, false, b1,
                                                      (short)0, acc1, false, false);
        acc2 = __builtin_amdgcn_wmma_f32_16x16x32_f16(false, a, false, b2,
                                                      (short)0, acc2, false, false);
        acc3 = __builtin_amdgcn_wmma_f32_16x16x32_f16(false, a, false, b3,
                                                      (short)0, acc3, false, false);
    }

    const int  g  = lane >> 4;
    const int  r  = lane & 15;
    float* o0 = out + ((long)mTile * 16 + g * 8) * OUTF + (long)nGroup * 64 + r;
#pragma unroll
    for (int p = 0; p < 8; ++p) {
        float* row = o0 + (long)p * OUTF;
        __builtin_nontemporal_store(acc0[p], row + 0);
        __builtin_nontemporal_store(acc1[p], row + 16);
        __builtin_nontemporal_store(acc2[p], row + 32);
        __builtin_nontemporal_store(acc3[p], row + 48);
    }
}

// ---------------------------------------------------------------------------
extern "C" void kernel_launch(void* const* d_in, const int* in_sizes, int n_in,
                              void* d_out, int out_size, void* d_ws, size_t ws_size,
                              hipStream_t stream) {
    (void)in_sizes; (void)n_in; (void)out_size;

    const float* x    = (const float*)d_in[0];   // [16384,16]
    const float* coef = (const float*)d_in[1];   // [16,4096,16]
    float*       out  = (float*)d_out;           // [16384,4096]

    const size_t bytesA = (size_t)BATCH * KDIM * sizeof(_Float16);  // 8 MB
    const size_t bytesB = (size_t)KDIM * OUTF * sizeof(_Float16);   // 2 MB
    if (ws_size < bytesA + bytesB) return;       // insufficient scratch

    _Float16* pA = (_Float16*)d_ws;
    _Float16* pB = (_Float16*)((char*)d_ws + bytesA);

    // 1) basis + pack A-fragments: 16384*16 threads
    kan_basis_pack<<<dim3((BATCH * INF) / 256), dim3(256), 0, stream>>>(x, pA);

    // 2) pack B-fragments: 16*4096 threads
    kan_coef_pack<<<dim3((INF * OUTF) / 256), dim3(256), 0, stream>>>(coef, pB);

    // 3) GEMM: 1024*64 waves, 8 waves per 256-thread block
    kan_gemm<<<dim3((BATCH / 16) * (OUTF / 64) / 8), dim3(256), 0, stream>>>(pA, pB, out);
}
